// RTGAE_59528246723272
// MI455X (gfx1250) — matrix operand order repaired and checked by reading, not compiled, not run on hardware
//
#include <hip/hip_runtime.h>

// ---------------------------------------------------------------------------
// RTGAE tree-GRU encode for MI455X (gfx1250, wave32, WMMA).
// GEMMs run as v_wmma_f32_16x16x32_bf16 (bf16 in / f32 accumulate).
// Each wave owns MT 16x16 output tiles stacked in M (B-fragment reuse).
// All row base addresses hoisted out of the K-loop -> immediate-offset loads.
// ---------------------------------------------------------------------------

typedef __attribute__((ext_vector_type(16))) __bf16 v16bf;
typedef __attribute__((ext_vector_type(8)))  float  v8f;

#define DIM   512
#define NSYM  64
#define NTOT  4681   // 1+8+64+512+4096
#define BATCH 32

// f32 -> bf16 via hardware convert (RNE, v_cvt_pk_bf16_f32)
__device__ __forceinline__ __bf16 f2bf(float f) { return (__bf16)f; }

// branch-free fast transcendentals (v_exp_f32 + v_rcp_f32)
__device__ __forceinline__ float fast_rcp(float x) { return __builtin_amdgcn_rcpf(x); }
__device__ __forceinline__ float sigm(float v) {
  return fast_rcp(1.0f + __expf(-v));
}
__device__ __forceinline__ float fast_tanh(float v) {
  return 2.0f * fast_rcp(1.0f + __expf(-2.0f * v)) - 1.0f;
}

__device__ __forceinline__ v8f wmma_bf16(v16bf a, v16bf b, v8f c) {
  // (neg_a, A, neg_b, B, c_mod, C, reuse_a, reuse_b)
  return __builtin_amdgcn_wmma_f32_16x16x32_bf16(false, a, false, b, (short)0, c,
                                                 false, false);
}

// A fragment from a hoisted per-lane row pointer.  Contiguous 16-bit A layout:
//   lanes 0-15 (m=lane): K = {0..15}
//   lanes 16-31        : K = {16..31}
// p already includes  (m-row)*lda + (lane>>4)*16 ;  caller adds kb*32.
__device__ __forceinline__ v16bf load_a8(const float* __restrict__ p) {
  v16bf out;
#pragma unroll
  for (int e = 0; e < 8; ++e) out[e] = f2bf(p[e]);
#pragma unroll
  for (int e = 0; e < 8; ++e) out[8 + e] = f2bf(p[8 + e]);
  return out;
}

// ---------------------------------------------------------------------------
// Fused GRU step.  out[m, :] = GRU(x[m], h[m]) for m in [0, M).
//  SIB  = true : gi = x @ Wih.T (GEMM);  false: gi = Wih_par[:, lab[m]] gather
//  HASH = true : h present (gh GEMM + z*h term); false: h == 0
//  MT          : 16-row tiles per wave stacked in M (B-fragment reuse)
// Block = 8 waves arranged 2(M) x 4(N): block tile = 32*MT rows x 64 cols.
// B layout: lanes 0-15 hold K=0..15 contiguous, lanes 16-31 K=16..31;
// B(k,n) = W[nrow0+n][kb*32+k]  (weights row-major -> K-contiguous 32B loads).
// ---------------------------------------------------------------------------
template <bool SIB, bool HASH, int MT>
__global__ __launch_bounds__(256)
void gru_step_kernel(const float* __restrict__ x, size_t x_lda,
                     const __bf16* __restrict__ Wih,
                     const __bf16* __restrict__ Whh,
                     const float* __restrict__ bih,
                     const float* __restrict__ bhh,
                     const float* __restrict__ h_in,
                     const float* __restrict__ Wih_par,
                     const int*   __restrict__ labels,
                     int nd_shift,
                     float* __restrict__ out) {
  const int lane = threadIdx.x & 31;
  const int wave = threadIdx.x >> 5;
  const int m0 = (blockIdx.x * 2 + (wave >> 2)) * (16 * MT);
  const int n0 = blockIdx.y * 64 + (wave & 3) * 16;
  const int lm = lane & 15;
  const int g  = lane >> 4;

  // ---- hoisted per-lane row base pointers (loop-invariant) ----
  const float  *xrow[MT], *hrow[MT];
  const __bf16 *bi[3], *bh[3];
  if (SIB) {
#pragma unroll
    for (int t = 0; t < MT; ++t)
      xrow[t] = x + (size_t)(m0 + 16 * t + lm) * x_lda + g * 16;
#pragma unroll
    for (int gt = 0; gt < 3; ++gt)
      bi[gt] = Wih + (size_t)(n0 + gt * DIM + lm) * DIM + g * 16;
  }
  if (HASH) {
#pragma unroll
    for (int t = 0; t < MT; ++t)
      hrow[t] = h_in + (size_t)(m0 + 16 * t + lm) * DIM + g * 16;
#pragma unroll
    for (int gt = 0; gt < 3; ++gt)
      bh[gt] = Whh + (size_t)(n0 + gt * DIM + lm) * DIM + g * 16;
  }

  v8f ai[3][MT] = {};   // gi accumulators (r,z,n)
  v8f ah[3][MT] = {};   // gh accumulators (r,z,n)

#pragma unroll
  for (int kb = 0; kb < DIM / 32; ++kb) {
    if (SIB) {
      v16bf a[MT];
#pragma unroll
      for (int t = 0; t < MT; ++t) a[t] = load_a8(xrow[t] + kb * 32);
#pragma unroll
      for (int gt = 0; gt < 3; ++gt) {
        const v16bf b = *(const v16bf*)(bi[gt] + kb * 32);
#pragma unroll
        for (int t = 0; t < MT; ++t) ai[gt][t] = wmma_bf16(a[t], b, ai[gt][t]);
      }
    }
    if (HASH) {
      v16bf a[MT];
#pragma unroll
      for (int t = 0; t < MT; ++t) a[t] = load_a8(hrow[t] + kb * 32);
#pragma unroll
      for (int gt = 0; gt < 3; ++gt) {
        const v16bf b = *(const v16bf*)(bh[gt] + kb * 32);
#pragma unroll
        for (int t = 0; t < MT; ++t) ah[gt][t] = wmma_bf16(a[t], b, ah[gt][t]);
      }
    }
  }

  // C/D layout: acc[v] holds m = r0 + v + 8*(lane>=16), n = n0 + (lane&15)
  const int col = n0 + lm;
  const float bir = bih[col], biz = bih[DIM + col], bin = bih[2 * DIM + col];
  const float bhr = bhh[col], bhz = bhh[DIM + col], bhn = bhh[2 * DIM + col];
  const int ndmask = (1 << nd_shift) - 1;

#pragma unroll
  for (int t = 0; t < MT; ++t) {
#pragma unroll
    for (int v = 0; v < 8; ++v) {
      const int m = m0 + 16 * t + v + 8 * g;
      float gir, giz, gin;
      if (SIB) {
        gir = ai[0][t][v] + bir; giz = ai[1][t][v] + biz; gin = ai[2][t][v] + bin;
      } else {
        const int lab = labels[(size_t)(m >> nd_shift) * NTOT + (m & ndmask)];
        gir = Wih_par[(size_t)col             * NSYM + lab] + bir;
        giz = Wih_par[(size_t)(DIM + col)     * NSYM + lab] + biz;
        gin = Wih_par[(size_t)(2 * DIM + col) * NSYM + lab] + bin;
      }
      const float ghr = (HASH ? ah[0][t][v] : 0.0f) + bhr;
      const float ghz = (HASH ? ah[1][t][v] : 0.0f) + bhz;
      const float ghn = (HASH ? ah[2][t][v] : 0.0f) + bhn;
      const float r  = sigm(gir + ghr);
      const float z  = sigm(giz + ghz);
      const float nn = fast_tanh(gin + r * ghn);
      const float ho = HASH ? h_in[(size_t)m * DIM + col] : 0.0f;
      out[(size_t)m * DIM + col] = (1.0f - z) * nn + z * ho;
    }
  }
}

// Leaf level (d=4): h = gru(onehot(lab), 0) -> pure gather + elementwise.
__global__ __launch_bounds__(256)
void leaf_kernel(const int* __restrict__ labels,      // already offset to level 4
                 const float* __restrict__ Wih_par,   // [1536, 64] f32
                 const float* __restrict__ bih,
                 const float* __restrict__ bhh,
                 float* __restrict__ out) {           // [B*4096, 512]
  const size_t idx = (size_t)blockIdx.x * 256 + threadIdx.x;  // < B*4096*512
  const int col = (int)(idx & (DIM - 1));
  const size_t mm = idx >> 9;          // b*4096 + p
  const int b = (int)(mm >> 12);
  const int p = (int)(mm & 4095);
  const int lab = labels[(size_t)b * NTOT + p];
  const float gir = Wih_par[(size_t)col             * NSYM + lab] + bih[col];
  const float giz = Wih_par[(size_t)(DIM + col)     * NSYM + lab] + bih[DIM + col];
  const float gin = Wih_par[(size_t)(2 * DIM + col) * NSYM + lab] + bih[2 * DIM + col];
  const float r  = sigm(gir + bhh[col]);
  const float z  = sigm(giz + bhh[DIM + col]);
  const float nn = fast_tanh(gin + r * bhh[2 * DIM + col]);
  out[idx] = (1.0f - z) * nn;
}

__global__ __launch_bounds__(256)
void cvt_bf16_kernel(const float* __restrict__ src, __bf16* __restrict__ dst, int n) {
  const int i = blockIdx.x * 256 + threadIdx.x;
  if (i < n) dst[i] = f2bf(src[i]);
}

// ---------------------------------------------------------------------------
// Host-side: one tree level = 8 sibling steps (ping-pong) + parent cell.
// ---------------------------------------------------------------------------
template <int MT>
static void run_level(dim3 grid, hipStream_t stream, const float* child,
                      const __bf16* wih_sib, const __bf16* whh_sib,
                      const float* bih_sib, const float* bhh_sib,
                      const __bf16* whh_par, const float* bih_par,
                      const float* bhh_par, const float* Wih_par,
                      const int* labels_lvl, int nd_shift,
                      float* C0, float* C1, float* out_lvl) {
  // a = 0: h == 0 (no zero-init needed)
  gru_step_kernel<true, false, MT><<<grid, 256, 0, stream>>>(
      child, (size_t)(8 * DIM), wih_sib, whh_sib, bih_sib, bhh_sib,
      nullptr, nullptr, nullptr, 0, C0);
  float* cur = C0; float* nxt = C1;
  for (int a = 1; a < 8; ++a) {
    gru_step_kernel<true, true, MT><<<grid, 256, 0, stream>>>(
        child + (size_t)a * DIM, (size_t)(8 * DIM), wih_sib, whh_sib,
        bih_sib, bhh_sib, cur, nullptr, nullptr, 0, nxt);
    float* t = cur; cur = nxt; nxt = t;
  }
  // parent cell: gi gathered from Wih_par via labels, gh = h_child @ Whh_par.T
  gru_step_kernel<false, true, MT><<<grid, 256, 0, stream>>>(
      nullptr, 0, nullptr, whh_par, bih_par, bhh_par,
      cur, Wih_par, labels_lvl, nd_shift, out_lvl);
}

extern "C" void kernel_launch(void* const* d_in, const int* in_sizes, int n_in,
                              void* d_out, int out_size, void* d_ws, size_t ws_size,
                              hipStream_t stream) {
  (void)in_sizes; (void)n_in; (void)out_size; (void)ws_size;

  const int*   labels  = (const int*)  d_in[0];
  const float* Wih_sib = (const float*)d_in[1];
  const float* Whh_sib = (const float*)d_in[2];
  const float* bih_sib = (const float*)d_in[3];
  const float* bhh_sib = (const float*)d_in[4];
  const float* Wih_par = (const float*)d_in[5];
  const float* Whh_par = (const float*)d_in[6];
  const float* bih_par = (const float*)d_in[7];
  const float* bhh_par = (const float*)d_in[8];

  // ---- workspace layout (bytes) ----
  char* ws = (char*)d_ws;
  const size_t WN  = (size_t)3 * DIM * DIM;        // 786432 elems per weight
  const size_t WBF = WN * 2;                       // 1,572,864 B (bf16)
  __bf16* wih_sib_bf = (__bf16*)(ws + 0 * WBF);
  __bf16* whh_sib_bf = (__bf16*)(ws + 1 * WBF);
  __bf16* whh_par_bf = (__bf16*)(ws + 2 * WBF);
  size_t off = 3 * WBF;                            // 4,718,592 (256B aligned)
  float* H4 = (float*)(ws + off); off += (size_t)BATCH * 4096 * DIM * 4;  // 256 MB
  float* H3 = (float*)(ws + off); off += (size_t)BATCH * 512  * DIM * 4;  // 32 MB
  float* H2 = (float*)(ws + off); off += (size_t)BATCH * 64   * DIM * 4;  //  4 MB
  float* H1 = (float*)(ws + off); off += (size_t)BATCH * 8    * DIM * 4;
  float* C0 = (float*)(ws + off); off += (size_t)BATCH * 512  * DIM * 4;  // scan state
  float* C1 = (float*)(ws + off);

  // ---- convert weights to bf16 (per launch; deterministic) ----
  {
    dim3 gcv((unsigned)((WN + 255) / 256));
    cvt_bf16_kernel<<<gcv, 256, 0, stream>>>(Wih_sib, wih_sib_bf, (int)WN);
    cvt_bf16_kernel<<<gcv, 256, 0, stream>>>(Whh_sib, whh_sib_bf, (int)WN);
    cvt_bf16_kernel<<<gcv, 256, 0, stream>>>(Whh_par, whh_par_bf, (int)WN);
  }

  // ---- level 4 (leaves): gather-only ----
  {
    const size_t total = (size_t)BATCH * 4096 * DIM;
    leaf_kernel<<<dim3((unsigned)(total / 256)), 256, 0, stream>>>(
        labels + 585, Wih_par, bih_par, bhh_par, H4);
  }

  float* Hlv[5] = {(float*)d_out, H1, H2, H3, H4};
  const int OFF[4] = {0, 1, 9, 73};

  // ---- levels 3..0: 8 sequential sibling GRU steps, then parent cell ----
  for (int d = 3; d >= 0; --d) {
    const int n_d = 1 << (3 * d);
    const int M = BATCH * n_d;
    const float* child = Hlv[d + 1];            // [M, 8, DIM]: row(m,a) = m*8+a
    if (M >= 64) {
      run_level<2>(dim3((unsigned)(M / 64), DIM / 64), stream, child,
                   wih_sib_bf, whh_sib_bf, bih_sib, bhh_sib,
                   whh_par_bf, bih_par, bhh_par, Wih_par,
                   labels + OFF[d], 3 * d, C0, C1, Hlv[d]);
    } else {  // level 0 only (M == 32)
      run_level<1>(dim3((unsigned)(M / 32), DIM / 64), stream, child,
                   wih_sib_bf, whh_sib_bf, bih_sib, bhh_sib,
                   whh_par_bf, bih_par, bhh_par, Wih_par,
                   labels + OFF[d], 3 * d, C0, C1, Hlv[d]);
    }
  }
}
